// QLoRABigNet_1546188227159
// MI455X (gfx1250) — compile-verified, matrix operand order checked by reading, not run on hardware
//
#include <hip/hip_runtime.h>

#define D 1024
#define R 8
#define GS 16
#define G (D / GS)
#define NLIN 18
#define ALPHA 4.0f

#define TILE_M 128
#define TILE_N 128
#define TILE_K 32
#define KPAD 16                      // LDS row padding (halves)
#define KSTEPS (D / TILE_K)

typedef __attribute__((ext_vector_type(16))) _Float16 v16h;
typedef __attribute__((ext_vector_type(8)))  _Float16 v8h;
typedef __attribute__((ext_vector_type(8)))  float    v8f;
typedef uint32_t u32x4 __attribute__((ext_vector_type(4)));
typedef uint32_t u32x8 __attribute__((ext_vector_type(8)));

// ---------------------------------------------------------------------------
// CDNA5 async / TDM helpers
// ---------------------------------------------------------------------------
__device__ __forceinline__ uint32_t lds_addr32(const void* p) {
    // Shared-aperture generic address keeps the LDS byte offset in addr[31:0]
    return (uint32_t)(uintptr_t)p;
}

__device__ __forceinline__ void async_load_b128(uint32_t lds, const void* gptr) {
    asm volatile("global_load_async_to_lds_b128 %0, %1, off"
                 :: "v"(lds), "v"((uint64_t)(uintptr_t)gptr) : "memory");
}

__device__ __forceinline__ void wait_asynccnt0() {
    asm volatile("s_wait_asynccnt 0x0" ::: "memory");
}

// TDM: DMA a TILE_N x TILE_K fp16 tile (row stride D halves) from global into
// LDS with 8-DWORD padding every 16 DWORDs (matches TILE_K+KPAD row stride).
__device__ __forceinline__ void tdm_load_wtile(uint32_t lds, const _Float16* gsrc) {
    const uint64_t ga = (uint64_t)(uintptr_t)gsrc;
    u32x4 g0;
    g0[0] = 1u;                                    // count=1, user-mode D#
    g0[1] = lds;                                   // lds_addr (bytes)
    g0[2] = (uint32_t)ga;                          // global_addr[31:0]
    g0[3] = (uint32_t)((ga >> 32) & 0x1FFFFFFu)    // global_addr[56:32]
            | (2u << 30);                          // type = 2 ("image")
    u32x8 g1;
    g1[0] = (1u << 16)                             // data_size = 2 bytes
          | (1u << 20)                             // pad_enable
          | (3u << 22)                             // pad_interval: every 16 DWORDs
          | (7u << 25);                            // pad_amount: 8 DWORDs
    g1[1] = ((uint32_t)D & 0xFFFFu) << 16;         // tensor_dim0[15:0]
    g1[2] = ((uint32_t)D & 0xFFFFu) << 16;         // dim0[31:16]=0 | tensor_dim1[15:0]
    g1[3] = ((uint32_t)TILE_K) << 16;              // dim1[31:16]=0 | tile_dim0
    g1[4] = (uint32_t)TILE_N;                      // tile_dim1 | tile_dim2=0
    g1[5] = (uint32_t)D;                           // tensor_dim0_stride[31:0]
    g1[6] = 0u;                                    // stride[47:32]=0 | dim1_stride[15:0]
    g1[7] = (uint32_t)((size_t)D * D >> 16);       // tensor_dim1_stride[47:16]
    asm volatile("tensor_load_to_lds %0, %1" :: "s"(g0), "s"(g1) : "memory");
}

// ---------------------------------------------------------------------------
// int4 group dequant -> fp16 weights (all 18 layers, [D][D] row-major)
// ---------------------------------------------------------------------------
__global__ __launch_bounds__(256) void dequant_kernel(const int* __restrict__ qw,
                                                      const _Float16* __restrict__ scales,
                                                      _Float16* __restrict__ W16) {
    int g = blockIdx.x * 256 + threadIdx.x;
    if (g >= NLIN * D * G) return;
    float s = (float)scales[g];
    const int* q = qw + (size_t)g * GS;
    _Float16* w = W16 + (size_t)g * GS;
#pragma unroll
    for (int t = 0; t < GS; ++t) w[t] = (_Float16)((float)(q[t] - 7) * s);
}

// ---------------------------------------------------------------------------
// h = x (fp32) and fp16 mirror
// ---------------------------------------------------------------------------
__global__ __launch_bounds__(256) void init_kernel(const float* __restrict__ x,
                                                   float* __restrict__ H,
                                                   _Float16* __restrict__ Xh, size_t n) {
    size_t i = (size_t)blockIdx.x * 256 + threadIdx.x;
    if (i < n) { float v = x[i]; H[i] = v; Xh[i] = (_Float16)v; }
}

// ---------------------------------------------------------------------------
// LoRA stage 1: T[n,r] = dot(act[n,:], la[r,:])  (fp32, rank 8)
// ---------------------------------------------------------------------------
__global__ __launch_bounds__(256) void lora_t_kernel(const float* __restrict__ act,
                                                     const float* __restrict__ la,
                                                     float* __restrict__ T, int Mrows) {
    size_t idx = (size_t)blockIdx.x * 256 + threadIdx.x;
    if (idx >= (size_t)Mrows * R) return;
    int n = (int)(idx >> 3);
    int r = (int)(idx & 7);
    const float4* xr = (const float4*)(act + (size_t)n * D);
    const float4* ar = (const float4*)(la + (size_t)r * D);
    float s = 0.0f;
    for (int k = 0; k < D / 4; ++k) {
        float4 a = xr[k], b = ar[k];
        s += a.x * b.x + a.y * b.y + a.z * b.z + a.w * b.w;
    }
    T[idx] = s;
}

// ---------------------------------------------------------------------------
// Main fused GEMM: Y = X(fp16) * W^T + bias + 4*(T * lb^T), optional ReLU.
// 128x128 block tile, 8 wave32 waves, each 32x64 (2x4 v_wmma_f32_16x16x32_f16).
// X tile: per-lane global_load_async_to_lds_b128 (ASYNCcnt).
// W tile: Tensor Data Mover tensor_load_to_lds, wave 0 only (TENSORcnt).
// Double buffered; DMA of tile kt+1 overlaps WMMA on tile kt.
// ---------------------------------------------------------------------------
__global__ __launch_bounds__(256) void qlora_gemm_kernel(
    const _Float16* __restrict__ X, const _Float16* __restrict__ W,
    const _Float16* __restrict__ bias, const float* __restrict__ T,
    const float* __restrict__ LB, float* __restrict__ Aout,
    _Float16* __restrict__ Xout, int relu) {

    __shared__ __align__(16) _Float16 Xs[2][TILE_M][TILE_K + KPAD];
    __shared__ __align__(16) _Float16 Ws[2][TILE_N][TILE_K + KPAD];
    __shared__ __align__(16) float Tsh[TILE_M][R];
    __shared__ __align__(16) float Lsh[TILE_N][R];

    const int tid  = threadIdx.x;
    const int lane = tid & 31;
    const int wave = tid >> 5;
    const int wm = wave >> 1;     // 0..3 -> 32-row stripes
    const int wn = wave & 1;      // 0..1 -> 64-col stripes
    const int m0 = blockIdx.y * TILE_M;
    const int n0 = blockIdx.x * TILE_N;

    const int lrow = tid >> 1;          // 0..127
    const int lcol = (tid & 1) * 16;    // 0 or 16 halves

    v8f acc[2][4] = {};

    const _Float16* xsrc = X + (size_t)(m0 + lrow) * D + lcol;

    auto stage_tiles = [&](int buf, int k0) {
        const uint32_t xa = lds_addr32(&Xs[buf][lrow][lcol]);
        async_load_b128(xa,      xsrc + k0);
        async_load_b128(xa + 16, xsrc + k0 + 8);
        if (wave == 0)
            tdm_load_wtile(lds_addr32(&Ws[buf][0][0]),
                           W + (size_t)n0 * D + k0);
    };

    stage_tiles(0, 0);
    {   // stage LoRA operands for the epilogue while the first tiles DMA in
        const int er = tid >> 1, ec = (tid & 1) * 4;
        *(float4*)&Tsh[er][ec] = *(const float4*)(T  + (size_t)(m0 + er) * R + ec);
        *(float4*)&Lsh[er][ec] = *(const float4*)(LB + (size_t)(n0 + er) * R + ec);
    }
    wait_asynccnt0();
    if (wave == 0) __builtin_amdgcn_s_wait_tensorcnt((short)0);
    __syncthreads();

    const int fr = lane & 15;        // M (A) / N (B) index within 16
    const int kb = (lane >> 4) * 8;  // K sub-offset per ISA fragment layout

    for (int kt = 0; kt < KSTEPS; ++kt) {
        const int buf = kt & 1;
        if (kt + 1 < KSTEPS) stage_tiles(buf ^ 1, (kt + 1) * TILE_K);
        if (kt + 2 < KSTEPS)
            __builtin_prefetch(xsrc + (kt + 2) * TILE_K, 0, 1);

        v16h afrag[2], bfrag[4];
#pragma unroll
        for (int mt = 0; mt < 2; ++mt) {
            const int row = wm * 32 + mt * 16 + fr;
            v8h lo = *(const v8h*)&Xs[buf][row][kb];
            v8h hi = *(const v8h*)&Xs[buf][row][kb + 16];
            afrag[mt] = __builtin_shufflevector(lo, hi,
                0, 1, 2, 3, 4, 5, 6, 7, 8, 9, 10, 11, 12, 13, 14, 15);
        }
#pragma unroll
        for (int nt = 0; nt < 4; ++nt) {
            const int col = wn * 64 + nt * 16 + fr;
            v8h lo = *(const v8h*)&Ws[buf][col][kb];
            v8h hi = *(const v8h*)&Ws[buf][col][kb + 16];
            bfrag[nt] = __builtin_shufflevector(lo, hi,
                0, 1, 2, 3, 4, 5, 6, 7, 8, 9, 10, 11, 12, 13, 14, 15);
        }
#pragma unroll
        for (int mt = 0; mt < 2; ++mt)
#pragma unroll
            for (int nt = 0; nt < 4; ++nt)
                acc[mt][nt] = __builtin_amdgcn_wmma_f32_16x16x32_f16(
                    false, afrag[mt], false, bfrag[nt],
                    (short)0, acc[mt][nt], false, false);

        wait_asynccnt0();
        if (wave == 0) __builtin_amdgcn_s_wait_tensorcnt((short)0);
        __syncthreads();
    }

    // Epilogue: bias + 4*(T . lb) + optional ReLU; per ISA C/D layout:
    // lanes 0-15: N=lane, M=r ; lanes 16-31: N=lane-16, M=r+8
#pragma unroll
    for (int mt = 0; mt < 2; ++mt) {
#pragma unroll
        for (int nt = 0; nt < 4; ++nt) {
            const int lncol = wn * 64 + nt * 16 + fr;
            const float bcol = (float)bias[n0 + lncol];
            const float4 b0 = *(const float4*)&Lsh[lncol][0];
            const float4 b1 = *(const float4*)&Lsh[lncol][4];
#pragma unroll
            for (int r = 0; r < 8; ++r) {
                const int lmrow = wm * 32 + mt * 16 + (lane >> 4) * 8 + r;
                const float4 t0 = *(const float4*)&Tsh[lmrow][0];
                const float4 t1 = *(const float4*)&Tsh[lmrow][4];
                float v = acc[mt][nt][r] + bcol +
                          ALPHA * (t0.x * b0.x + t0.y * b0.y + t0.z * b0.z + t0.w * b0.w +
                                   t1.x * b1.x + t1.y * b1.y + t1.z * b1.z + t1.w * b1.w);
                if (relu) v = fmaxf(v, 0.0f);
                const size_t o = (size_t)(m0 + lmrow) * D + (n0 + lncol);
                Aout[o] = v;
                Xout[o] = (_Float16)v;
            }
        }
    }
}

// ---------------------------------------------------------------------------
// h = LN(y + h) (or just y + h on last block); refresh fp16 mirror
// ---------------------------------------------------------------------------
__global__ __launch_bounds__(256) void ln_res_kernel(const float* __restrict__ Ain,
                                                     const float* __restrict__ Hin,
                                                     const float* __restrict__ gamma,
                                                     const float* __restrict__ beta,
                                                     float* __restrict__ Hout,
                                                     _Float16* __restrict__ Xout, int doLN) {
    __shared__ float ssum[256], ssq[256];
    const int row = blockIdx.x;
    const int tid = threadIdx.x;
    float v[4];
    float s = 0.0f, q = 0.0f;
#pragma unroll
    for (int i = 0; i < 4; ++i) {
        const int c = tid + i * 256;
        const size_t o = (size_t)row * D + c;
        const float x = Ain[o] + Hin[o];
        v[i] = x; s += x; q += x * x;
    }
    if (!doLN) {
#pragma unroll
        for (int i = 0; i < 4; ++i) Hout[(size_t)row * D + tid + i * 256] = v[i];
        return;
    }
    ssum[tid] = s; ssq[tid] = q;
    __syncthreads();
    for (int off = 128; off > 0; off >>= 1) {
        if (tid < off) { ssum[tid] += ssum[tid + off]; ssq[tid] += ssq[tid + off]; }
        __syncthreads();
    }
    const float mean = ssum[0] * (1.0f / D);
    const float var  = ssq[0] * (1.0f / D) - mean * mean;
    const float rs   = rsqrtf(var + 1e-5f);
#pragma unroll
    for (int i = 0; i < 4; ++i) {
        const int c = tid + i * 256;
        const float y = (v[i] - mean) * rs * gamma[c] + beta[c];
        const size_t o = (size_t)row * D + c;
        Hout[o] = y;
        Xout[o] = (_Float16)y;
    }
}

// ---------------------------------------------------------------------------
extern "C" void kernel_launch(void* const* d_in, const int* in_sizes, int n_in,
                              void* d_out, int out_size, void* d_ws, size_t ws_size,
                              hipStream_t stream) {
    (void)n_in; (void)out_size; (void)ws_size;
    const float*    x      = (const float*)d_in[0];
    const int*      qw     = (const int*)d_in[1];
    const _Float16* scales = (const _Float16*)d_in[2];
    const _Float16* biases = (const _Float16*)d_in[3];
    const float*    la     = (const float*)d_in[4];
    const float*    lb     = (const float*)d_in[5];
    const float*    gamma  = (const float*)d_in[6];
    const float*    beta   = (const float*)d_in[7];

    const int Mrows = in_sizes[0] / D;
    const size_t nelem = (size_t)Mrows * D;

    uintptr_t p = (uintptr_t)d_ws;
    auto take = [&](size_t bytes) -> void* {
        void* r = (void*)p;
        p += (bytes + 255) & ~(size_t)255;
        return r;
    };
    _Float16* W16 = (_Float16*)take((size_t)NLIN * D * D * sizeof(_Float16));
    float*    H   = (float*)take(nelem * sizeof(float));
    float*    A0  = (float*)take(nelem * sizeof(float));
    float*    A1  = (float*)take(nelem * sizeof(float));
    _Float16* Xh0 = (_Float16*)take(nelem * sizeof(_Float16));
    _Float16* Xh1 = (_Float16*)take(nelem * sizeof(_Float16));
    float*    T   = (float*)take((size_t)Mrows * R * sizeof(float));

    dequant_kernel<<<(NLIN * D * G + 255) / 256, 256, 0, stream>>>(qw, scales, W16);
    init_kernel<<<(unsigned)((nelem + 255) / 256), 256, 0, stream>>>(x, H, Xh0, nelem);

    float*    jinF[3]  = {H,  A0, A1};
    float*    joutF[3] = {A0, A1, A0};
    _Float16* jinH[3]  = {Xh0, Xh1, Xh0};
    _Float16* joutH[3] = {Xh1, Xh0, Xh1};

    const dim3 ggrid(D / TILE_N, Mrows / TILE_M);
    for (int blk = 0; blk < 6; ++blk) {
        for (int j = 0; j < 3; ++j) {
            const int i = blk * 3 + j;
            lora_t_kernel<<<(unsigned)(((size_t)Mrows * R + 255) / 256), 256, 0, stream>>>(
                jinF[j], la + (size_t)i * R * D, T, Mrows);
            qlora_gemm_kernel<<<ggrid, 256, 0, stream>>>(
                jinH[j], W16 + (size_t)i * D * D, biases + (size_t)i * D, T,
                lb + (size_t)i * D * R, joutF[j], joutH[j], j < 2 ? 1 : 0);
        }
        const int last = (blk == 5);
        const int g = blk < 5 ? blk : 4;
        ln_res_kernel<<<Mrows, 256, 0, stream>>>(
            A0, H, gamma + (size_t)g * D, beta + (size_t)g * D,
            last ? (float*)d_out : H, Xh0, last ? 0 : 1);
    }
}
